// AudioEncoder_57191784513658
// MI455X (gfx1250) — matrix operand order repaired and checked by reading, not compiled, not run on hardware
//
#include <hip/hip_runtime.h>
#include <hip/hip_bf16.h>
#include <math.h>
#include <stdint.h>

typedef __attribute__((ext_vector_type(16))) __bf16 v16bf;
typedef __attribute__((ext_vector_type(8)))  __bf16 v8bf;
typedef __attribute__((ext_vector_type(8)))  float  v8f;

#define TPB 256
#define BM  128
#define BN  128
#define KT  32
#define LDK 40      // padded K stride in LDS (elements): 80B rows, 16B-aligned frags

// CDNA5 direct global->LDS async copy path (ASYNCcnt). Flip to 0 if toolchain
// rejects the mnemonic; the fallback stages through VGPRs.
#define ASYNC_LDS 1

// Split workgroup barrier (CDNA5): drain LDS/async writes, then signal; waves
// issue the next tile's global gather BEFORE waiting, hiding barrier latency.
__device__ __forceinline__ void wg_signal() {
  asm volatile("s_wait_dscnt 0x0" ::: "memory");
#if ASYNC_LDS
  asm volatile("s_wait_asynccnt 0x0" ::: "memory");
#endif
  asm volatile("s_barrier_signal -1" ::: "memory");
}
__device__ __forceinline__ void wg_wait() {
  asm volatile("s_barrier_wait -1" ::: "memory");
}

// ---------------------------------------------------------------------------
// fp32 -> bf16 elementwise (grid-stride)
// ---------------------------------------------------------------------------
__global__ void k_to_bf16(const float* __restrict__ x, __bf16* __restrict__ y, size_t n) {
  size_t i = (size_t)blockIdx.x * blockDim.x + threadIdx.x;
  size_t st = (size_t)gridDim.x * blockDim.x;
  for (; i < n; i += st) y[i] = (__bf16)x[i];
}

// ---------------------------------------------------------------------------
// conv0: 1 -> 256 channels, K=10, stride 5, pad 5. Weights cached in LDS.
// ---------------------------------------------------------------------------
__global__ __launch_bounds__(TPB)
void k_conv0(const float* __restrict__ wav, const float* __restrict__ w,
             const float* __restrict__ bias, float* __restrict__ y,
             int Lin, int Lout) {
  __shared__ float ws0[2560];
  __shared__ float bs0[256];
  int tid = threadIdx.x;
  for (int i = tid; i < 2560; i += TPB) ws0[i] = w[i];
  bs0[tid] = bias[tid];
  __syncthreads();
  int idx = blockIdx.x * TPB + tid;
  if (idx >= 8 * Lout) return;
  int b = idx / Lout, l = idx - b * Lout;
  float in[10];
  int p0 = 5 * l - 5;
#pragma unroll
  for (int k = 0; k < 10; ++k) {
    int p = p0 + k;
    in[k] = (p >= 0 && p < Lin) ? wav[(size_t)b * Lin + p] : 0.f;
  }
  for (int oc = 0; oc < 256; ++oc) {
    float a = bs0[oc];
#pragma unroll
    for (int k = 0; k < 10; ++k) a = fmaf(ws0[oc * 10 + k], in[k], a);
    y[((size_t)b * 256 + oc) * Lout + l] = a;
  }
}

// ---------------------------------------------------------------------------
// GroupNorm stats: one block per (group, batch) -> mean, rstd
// ---------------------------------------------------------------------------
__global__ __launch_bounds__(TPB)
void k_gn_stats(const float* __restrict__ x, float* __restrict__ stats,
                int C, int L, int G) {
  int g = blockIdx.x, b = blockIdx.y;
  int cpg = C / G;
  size_t base = ((size_t)b * C + (size_t)g * cpg) * (size_t)L;
  size_t cnt = (size_t)cpg * (size_t)L;
  float s = 0.f, s2 = 0.f;
  for (size_t i = threadIdx.x; i < cnt; i += TPB) {
    float v = x[base + i];
    s += v; s2 += v * v;
  }
  __shared__ float sh[TPB], sh2[TPB];
  int tid = threadIdx.x;
  sh[tid] = s; sh2[tid] = s2;
  __syncthreads();
  for (int st = TPB / 2; st > 0; st >>= 1) {
    if (tid < st) { sh[tid] += sh[tid + st]; sh2[tid] += sh2[tid + st]; }
    __syncthreads();
  }
  if (tid == 0) {
    float mean = sh[0] / (float)cnt;
    float var  = sh2[0] / (float)cnt - mean * mean;
    stats[(b * G + g) * 2 + 0] = mean;
    stats[(b * G + g) * 2 + 1] = rsqrtf(var + 1e-5f);
  }
}

// ---------------------------------------------------------------------------
// GroupNorm apply + exact GELU + bf16 pack. grid.y = b*C + c
// ---------------------------------------------------------------------------
__global__ __launch_bounds__(TPB)
void k_gn_apply_gelu(const float* __restrict__ x, __bf16* __restrict__ y,
                     const float* __restrict__ stats,
                     const float* __restrict__ gamma, const float* __restrict__ beta,
                     int C, int L, int G) {
  int bc = blockIdx.y;
  int b = bc / C, c = bc - b * C;
  int cpg = C / G;
  int g = c / cpg;
  float mean = stats[(b * G + g) * 2 + 0];
  float rstd = stats[(b * G + g) * 2 + 1];
  float ga = gamma[c], be = beta[c];
  int l = blockIdx.x * TPB + threadIdx.x;
  if (l >= L) return;
  size_t i = ((size_t)b * C + c) * (size_t)L + l;
  float v = (x[i] - mean) * rstd * ga + be;
  v = 0.5f * v * (1.f + erff(v * 0.70710678118654752f));
  y[i] = (__bf16)v;
}

// ---------------------------------------------------------------------------
// Double-buffered, split-barrier-pipelined bf16 WMMA GEMM. D(f32) = A(MxK)*B(KxN)
//  MODE 0: conv im2col  B[k][n] = act[b, k/10, 2*(n0+n)+k%10-5]; Out=[b][M][N]+bias
//  MODE 1: projection   B[k][n] = act[n/Lin, k, n%Lin];          Out[n][M]=d+bias[m]
//  MODE 2: VQ scores    B[k][n] = featsbf[n*K+k];                Out[n][M]=extra[m]-2d
// Block: 256 thr = 8 waves (4x2); block tile 128x128; wave tile 32x64 (2x4 frags).
// A tile: global->LDS async copy (ASYNCcnt). B tile: gathered to regs *before*
// s_barrier_wait, committed as ds_store_b128 after WMMA on the other buffer.
// ---------------------------------------------------------------------------
template <int MODE>
__global__ __launch_bounds__(TPB)
void k_gemm_wmma(const __bf16* __restrict__ A, const __bf16* __restrict__ Bsrc,
                 float* __restrict__ Out, const float* __restrict__ extra,
                 int M, int Kdim, int N, int IC, int Lin) {
  __shared__ __bf16 As[2][BM][LDK];
  __shared__ __bf16 Bs[2][BN][LDK];

  const int tid  = threadIdx.x;
  const int lane = tid & 31;
  const int wave = tid >> 5;
  const int wm = wave & 3;           // M sub-tile (32 rows)
  const int wn = wave >> 2;          // N sub-tile (64 cols)
  const int n0 = blockIdx.x * BN;
  const int m0 = blockIdx.y * BM;
  const int b  = blockIdx.z;

  const v8f vzero = {0.f, 0.f, 0.f, 0.f, 0.f, 0.f, 0.f, 0.f};
  v8f acc[2][4];
#pragma unroll
  for (int i = 0; i < 2; ++i)
#pragma unroll
    for (int j = 0; j < 4; ++j) acc[i][j] = vzero;

  // ISA fragment geometry (wave32)
  const int arow = lane & 15;
  const int ak0  = (lane >> 4) << 3;   // A: K base 0/8 (second half at +16)
  const int bcol = lane & 15;
  const int bk0  = (lane >> 4) << 3;   // B: K base 0/8 (second half at +16)

  // per-thread copy coordinates (each thread moves 16 contiguous elements)
  const int cp_row = tid >> 1;         // 0..127
  const int cp_col = (tid & 1) << 4;   // 0 / 16

  const int nt = Kdim / KT;

#if !ASYNC_LDS
  v8bf apip0, apip1;
#endif
  __bf16 bpip[16];

  // ---- issue A tile (global -> LDS async, or regs in fallback)
  auto issueA = [&](int k0, int buf) {
    const __bf16* src = A + (size_t)(m0 + cp_row) * Kdim + (k0 + cp_col);
#if ASYNC_LDS
    unsigned dst = (unsigned)(uintptr_t)&As[buf][cp_row][cp_col];
    asm volatile("global_load_async_to_lds_b128 %0, %1, off"
                 :: "v"(dst), "v"(src) : "memory");
    asm volatile("global_load_async_to_lds_b128 %0, %1, off"
                 :: "v"(dst + 16u), "v"(src + 8) : "memory");
#else
    apip0 = *(const v8bf*)(src);
    apip1 = *(const v8bf*)(src + 8);
#endif
  };
  auto commitA = [&](int buf) {
#if !ASYNC_LDS
    *(v8bf*)(&As[buf][cp_row][cp_col])     = apip0;
    *(v8bf*)(&As[buf][cp_row][cp_col + 8]) = apip1;
#else
    (void)buf;
#endif
  };

  // ---- gather B tile slice into regs (16 contiguous k for one n)
  auto issueB = [&](int k0) {
    const int ng = n0 + cp_row;
    const bool okn = (ng < N);
    if (MODE == 2) {
      if (okn) {
        const __bf16* s = Bsrc + (size_t)ng * Kdim + (k0 + cp_col);
        v8bf l0 = *(const v8bf*)(s);
        v8bf l1 = *(const v8bf*)(s + 8);
#pragma unroll
        for (int i = 0; i < 8; ++i) { bpip[i] = l0[i]; bpip[8 + i] = l1[i]; }
      } else {
#pragma unroll
        for (int i = 0; i < 16; ++i) bpip[i] = (__bf16)0.f;
      }
    } else if (MODE == 1) {
      int nb = 0, nl = 0;
      if (okn) { nb = ng / Lin; nl = ng - nb * Lin; }
      size_t base = ((size_t)nb * IC) * (size_t)Lin + nl;
#pragma unroll
      for (int e = 0; e < 16; ++e) {
        int kg = k0 + cp_col + e;
        bpip[e] = okn ? Bsrc[base + (size_t)kg * Lin] : (__bf16)0.f;
      }
    } else {  // MODE 0: conv im2col
#pragma unroll
      for (int e = 0; e < 16; ++e) {
        int kg = k0 + cp_col + e;
        unsigned ic = (unsigned)kg / 10u;
        int kk = kg - (int)(ic * 10u);
        int pos = 2 * ng + kk - 5;
        __bf16 v = (__bf16)0.f;
        if (okn && pos >= 0 && pos < Lin)
          v = Bsrc[((size_t)b * IC + ic) * (size_t)Lin + pos];
        bpip[e] = v;
      }
    }
  };
  auto commitB = [&](int buf) {
    v8bf p0, p1;
#pragma unroll
    for (int i = 0; i < 8; ++i) { p0[i] = bpip[i]; p1[i] = bpip[8 + i]; }
    *(v8bf*)(&Bs[buf][cp_row][cp_col])     = p0;
    *(v8bf*)(&Bs[buf][cp_row][cp_col + 8]) = p1;
  };

  // ---- WMMA on one LDS buffer
  auto compute = [&](int buf) {
    v16bf af[2], bfr[4];
#pragma unroll
    for (int im = 0; im < 2; ++im) {
      const __bf16* p = &As[buf][wm * 32 + im * 16 + arow][ak0];
      v8bf lo = *(const v8bf*)(p);
      v8bf hi = *(const v8bf*)(p + 16);
      af[im] = __builtin_shufflevector(lo, hi, 0, 1, 2, 3, 4, 5, 6, 7,
                                               8, 9, 10, 11, 12, 13, 14, 15);
    }
#pragma unroll
    for (int in = 0; in < 4; ++in) {
      const __bf16* p = &Bs[buf][wn * 64 + in * 16 + bcol][bk0];
      v8bf lo = *(const v8bf*)(p);
      v8bf hi = *(const v8bf*)(p + 16);
      bfr[in] = __builtin_shufflevector(lo, hi, 0, 1, 2, 3, 4, 5, 6, 7,
                                                8, 9, 10, 11, 12, 13, 14, 15);
    }
#pragma unroll
    for (int im = 0; im < 2; ++im)
#pragma unroll
      for (int in = 0; in < 4; ++in)
        acc[im][in] = __builtin_amdgcn_wmma_f32_16x16x32_bf16(
            false, af[im], false, bfr[in], (short)0, acc[im][in], false, false);
  };

  // ---- split-barrier pipelined main loop
  // Round pairing: prologue signal + one wait per iteration + one signal per
  // non-final iteration (balanced). Writes to a buffer happen only after the
  // wait proving all waves finished reading it in the previous round.
  issueA(0, 0);
  issueB(0);
  commitA(0);
  commitB(0);
  wg_signal();                               // drains dscnt/asynccnt first

  for (int t = 0; t < nt; ++t) {
    const int cur = t & 1;
    const int nxt = cur ^ 1;
    const bool more = (t + 1 < nt);
    if (more) issueB((t + 1) * KT);          // global gather BEFORE barrier wait
    wg_wait();                               // buffer `cur` ready everywhere
    if (more) issueA((t + 1) * KT, nxt);     // async copy into `nxt` (safe now)
    compute(cur);
    if (more) {
      commitA(nxt);
      commitB(nxt);
      wg_signal();                           // publish `nxt` for next round
    }
  }

  // ---- epilogue (C/D layout: n = lane&15, m = 8*(lane>>4) + vgpr index)
  const int rowoff = (lane >> 4) << 3;
#pragma unroll
  for (int im = 0; im < 2; ++im) {
#pragma unroll
    for (int in = 0; in < 4; ++in) {
      int mbase = m0 + wm * 32 + im * 16 + rowoff;
      int n = n0 + wn * 64 + in * 16 + (lane & 15);
      if (n >= N) continue;
      if (MODE == 0) {
#pragma unroll
        for (int r = 0; r < 8; ++r)
          Out[((size_t)b * M + (mbase + r)) * (size_t)N + n] =
              acc[im][in][r] + extra[mbase + r];
      } else if (MODE == 1) {
#pragma unroll
        for (int r = 0; r < 8; ++r)
          Out[(size_t)n * M + (mbase + r)] = acc[im][in][r] + extra[mbase + r];
      } else {
#pragma unroll
        for (int r = 0; r < 8; ++r)
          Out[(size_t)n * M + (mbase + r)] = extra[mbase + r] - 2.f * acc[im][in][r];
      }
    }
  }
}

// ---------------------------------------------------------------------------
// codebook squared norms (fp32, full precision)
// ---------------------------------------------------------------------------
__global__ void k_code_norms(const float* __restrict__ cb, float* __restrict__ nrm,
                             int NC, int E) {
  int c = blockIdx.x * blockDim.x + threadIdx.x;
  if (c >= NC) return;
  float s = 0.f;
  for (int e = 0; e < E; ++e) { float v = cb[(size_t)c * E + e]; s = fmaf(v, v, s); }
  nrm[c] = s;
}

// ---------------------------------------------------------------------------
// per-row argmin over NC scores (ties -> lowest index); also writes mask = 1
// ---------------------------------------------------------------------------
__global__ __launch_bounds__(TPB)
void k_argmin(const float* __restrict__ scores, int* __restrict__ idx_out,
              int* __restrict__ mask_out, int NC) {
  int row = blockIdx.x;
  const float* s = scores + (size_t)row * NC;
  float best = 3.4e38f;
  int bi = 0x7fffffff;
  for (int c = threadIdx.x; c < NC; c += TPB) {
    float v = s[c];
    if (v < best) { best = v; bi = c; }
  }
  __shared__ float sv[TPB];
  __shared__ int   si[TPB];
  int tid = threadIdx.x;
  sv[tid] = best; si[tid] = bi;
  __syncthreads();
  for (int st = TPB / 2; st > 0; st >>= 1) {
    if (tid < st) {
      float v2 = sv[tid + st]; int i2 = si[tid + st];
      if (v2 < sv[tid] || (v2 == sv[tid] && i2 < si[tid])) { sv[tid] = v2; si[tid] = i2; }
    }
    __syncthreads();
  }
  if (tid == 0) { idx_out[row] = si[0]; mask_out[row] = 1; }
}

// ---------------------------------------------------------------------------
// emb = LayerNorm(codebook[idx] + pos_enc[l] + modality) * g + b   (E = 1536)
// ---------------------------------------------------------------------------
__global__ __launch_bounds__(TPB)
void k_emb_ln(const float* __restrict__ cb, const int* __restrict__ idx,
              const float* __restrict__ pos, const float* __restrict__ mod,
              const float* __restrict__ g, const float* __restrict__ bb,
              float* __restrict__ emb, int S, int E) {
  int row = blockIdx.x;
  int l = row % S;
  const float* code = cb + (size_t)idx[row] * E;
  const float* pe = pos + (size_t)l * E;
  int tid = threadIdx.x;
  float q[6];
  float s = 0.f, s2 = 0.f;
#pragma unroll
  for (int j = 0; j < 6; ++j) {
    int e = tid + j * TPB;
    float v = code[e] + pe[e] + mod[e];
    q[j] = v; s += v; s2 += v * v;
  }
  __shared__ float sh[TPB], sh2[TPB];
  sh[tid] = s; sh2[tid] = s2;
  __syncthreads();
  for (int st = TPB / 2; st > 0; st >>= 1) {
    if (tid < st) { sh[tid] += sh[tid + st]; sh2[tid] += sh2[tid + st]; }
    __syncthreads();
  }
  float mean = sh[0] / (float)E;
  float var  = sh2[0] / (float)E - mean * mean;
  float rstd = rsqrtf(var + 1e-5f);
#pragma unroll
  for (int j = 0; j < 6; ++j) {
    int e = tid + j * TPB;
    emb[(size_t)row * E + e] = (q[j] - mean) * rstd * g[e] + bb[e];
  }
}

// ---------------------------------------------------------------------------
extern "C" void kernel_launch(void* const* d_in, const int* in_sizes, int n_in,
                              void* d_out, int out_size, void* d_ws, size_t ws_size,
                              hipStream_t stream) {
  (void)in_sizes; (void)n_in; (void)out_size; (void)ws_size;

  const float* wav   = (const float*)d_in[0];
  const float* w0    = (const float*)d_in[1];
  const float* b0    = (const float*)d_in[2];
  const float* w1    = (const float*)d_in[3];
  const float* b1    = (const float*)d_in[4];
  const float* wr    = (const float*)d_in[5];
  const float* br    = (const float*)d_in[6];
  const float* gng0  = (const float*)d_in[7];
  const float* gnb0  = (const float*)d_in[8];
  const float* gngr  = (const float*)d_in[9];
  const float* gnbr  = (const float*)d_in[10];
  const float* projw = (const float*)d_in[11];
  const float* projb = (const float*)d_in[12];
  const float* cb    = (const float*)d_in[13];
  const float* pos   = (const float*)d_in[14];
  const float* mod   = (const float*)d_in[15];
  const float* lng   = (const float*)d_in[16];
  const float* lnb   = (const float*)d_in[17];

  const int B = 8;
  const int Lw = 312000;
  const int Lv[7] = {62401, 31201, 15601, 7801, 3901, 1951, 976};
  const int S = 976, E = 1536, NC = 1024;
  const int NR = B * S;  // 7808

  // ---- workspace carve-up
  char* ws = (char*)d_ws;
  size_t off = 0;
  auto take = [&](size_t bytes) -> char* {
    char* p = ws + off;
    off = (off + bytes + 255) & ~(size_t)255;
    return p;
  };
  const size_t NBIG = (size_t)B * 512 * 31201;  // covers every conv intermediate
  float*  convf  = (float*) take(NBIG * 4);
  __bf16* actA   = (__bf16*)take(NBIG * 2);
  __bf16* actB   = (__bf16*)take(NBIG * 2);
  __bf16* wbf    = (__bf16*)take((size_t)512 * 512 * 10 * 2);
  float*  stats  = (float*) take((size_t)B * 32 * 2 * 4);
  float*  feats  = (float*) take((size_t)NR * E * 4);
  __bf16* fbf    = (__bf16*)take((size_t)NR * E * 2);
  __bf16* pwbf   = (__bf16*)take((size_t)E * 512 * 2);
  __bf16* cbbf   = (__bf16*)take((size_t)NC * E * 2);
  float*  cnorm  = (float*) take((size_t)NC * 4);
  float*  scores = (float*) take((size_t)NR * NC * 4);

  float* emb_out  = (float*)d_out;
  int*   mask_out = (int*)(emb_out + (size_t)NR * E);
  int*   idx_out  = mask_out + NR;

  // ---- conv0 + GN(16) + GELU -> bf16
  {
    int total = B * Lv[0];
    k_conv0<<<(total + TPB - 1) / TPB, TPB, 0, stream>>>(wav, w0, b0, convf, Lw, Lv[0]);
    k_gn_stats<<<dim3(16, B), TPB, 0, stream>>>(convf, stats, 256, Lv[0], 16);
    k_gn_apply_gelu<<<dim3((Lv[0] + TPB - 1) / TPB, B * 256), TPB, 0, stream>>>(
        convf, actA, stats, gng0, gnb0, 256, Lv[0], 16);
  }

  // ---- conv1..conv6 as WMMA GEMMs + GN(32) + GELU -> bf16 (ping-pong)
  __bf16* cur = actA;
  __bf16* nxt = actB;
  for (int li = 1; li <= 6; ++li) {
    int Lin  = Lv[li - 1];
    int Lout = Lv[li];
    int IC   = (li == 1) ? 256 : 512;
    int Kd   = IC * 10;
    const float* wsrc = (li == 1) ? w1 : (wr + (size_t)(li - 2) * 512 * 512 * 10);
    const float* bsrc = (li == 1) ? b1 : (br + (size_t)(li - 2) * 512);
    const float* gga  = gngr + (size_t)(li - 1) * 512;
    const float* gbe  = gnbr + (size_t)(li - 1) * 512;

    k_to_bf16<<<1024, TPB, 0, stream>>>(wsrc, wbf, (size_t)512 * Kd);
    dim3 grid((Lout + BN - 1) / BN, 512 / BM, B);
    k_gemm_wmma<0><<<grid, TPB, 0, stream>>>(wbf, cur, convf, bsrc, 512, Kd, Lout, IC, Lin);
    k_gn_stats<<<dim3(32, B), TPB, 0, stream>>>(convf, stats, 512, Lout, 32);
    k_gn_apply_gelu<<<dim3((Lout + TPB - 1) / TPB, B * 512), TPB, 0, stream>>>(
        convf, nxt, stats, gga, gbe, 512, Lout, 32);
    __bf16* t = cur; cur = nxt; nxt = t;
  }
  // `cur` holds final activations [B][512][976] in bf16

  // ---- projection: feats[row][E] = x^T @ proj_w^T + proj_b
  k_to_bf16<<<512, TPB, 0, stream>>>(projw, pwbf, (size_t)E * 512);
  {
    dim3 grid((NR + BN - 1) / BN, E / BM, 1);
    k_gemm_wmma<1><<<grid, TPB, 0, stream>>>(pwbf, cur, feats, projb, E, 512, NR, 512, S);
  }

  // ---- VQ: scores = ||c||^2 - 2 * feats @ codebook^T, then argmin
  k_to_bf16<<<2048, TPB, 0, stream>>>(feats, fbf, (size_t)NR * E);
  k_to_bf16<<<1024, TPB, 0, stream>>>(cb, cbbf, (size_t)NC * E);
  k_code_norms<<<(NC + TPB - 1) / TPB, TPB, 0, stream>>>(cb, cnorm, NC, E);
  {
    dim3 grid((NR + BN - 1) / BN, NC / BM, 1);
    k_gemm_wmma<2><<<grid, TPB, 0, stream>>>(cbbf, fbf, scores, cnorm, NC, E, NR, 1, 1);
  }
  k_argmin<<<NR, TPB, 0, stream>>>(scores, idx_out, mask_out, NC);

  // ---- final: gather code + pos + modality, LayerNorm -> emb
  k_emb_ln<<<NR, TPB, 0, stream>>>(cb, idx_out, pos, mod, lng, lnb, emb_out, S, E);
}
